// CausalSelfAttention_13348758356006
// MI455X (gfx1250) — matrix operand order repaired and checked by reading, not compiled, not run on hardware
//
#include <hip/hip_runtime.h>

// ---------------------------------------------------------------------------
// Causal self-attention for MI455X (gfx1250, wave32, WMMA bf16 path)
// H=1024, NH=16, HD=64, B=2, S=2048, M=B*S=4096
// ---------------------------------------------------------------------------

#define H     1024
#define NH    16
#define HD    64
#define BATCH 2
#define SEQ   2048
#define MTOT  (BATCH * SEQ)   // 4096

typedef __attribute__((ext_vector_type(16))) __bf16 v16bf;
typedef __attribute__((ext_vector_type(8)))  __bf16 v8bf;
typedef __attribute__((ext_vector_type(8)))  float  v8f;

#define DEVINL __device__ __forceinline__

union V16U { v16bf v; v8bf h[2]; };

// A-fragment (16x32 bf16, row-major source, per-lane: 8 K @ +0, 8 K @ +16)
DEVINL v16bf ld_afrag(const __bf16* p) {
    V16U u;
    u.h[0] = *(const v8bf*)(p);
    u.h[1] = *(const v8bf*)(p + 16);
    return u.v;
}
// B-fragment (32x16 bf16, source stored [N,K] row-major, per-lane: 8 K @ +0, 8 K @ +16)
DEVINL v16bf ld_bfrag(const __bf16* p) {
    V16U u;
    u.h[0] = *(const v8bf*)(p);
    u.h[1] = *(const v8bf*)(p + 16);
    return u.v;
}

DEVINL v8f wmma_bf16(v16bf a, v16bf b, v8f c) {
    return __builtin_amdgcn_wmma_f32_16x16x32_bf16(
        /*neg_a=*/false, a, /*neg_b=*/false, b,
        /*c_mod=*/(short)0, c, /*reuse_a=*/false, /*reuse_b=*/false);
}

// XOR-lane shuffle via ds_swizzle (group-of-32 mode: and=0x1f, xor=M).
// M in {1,2,4,8} stays within each 16-lane half (C-layout row groups).
template<int M>
DEVINL float swz_xor(float x) {
    int r = __builtin_amdgcn_ds_swizzle(__builtin_bit_cast(int, x),
                                        0x1f | (M << 10));
    return __builtin_bit_cast(float, r);
}

// ---------------------------------------------------------------------------
// fp32 -> bf16 elementwise convert (hidden states)
// ---------------------------------------------------------------------------
__global__ __launch_bounds__(256) void cvt_f32_bf16(const float* __restrict__ in,
                                                    __bf16* __restrict__ out, int n) {
    int i = (blockIdx.x * 256 + threadIdx.x) * 4;
    if (i + 3 < n) {
        float4 v = *(const float4*)(in + i);
        out[i + 0] = (__bf16)v.x;
        out[i + 1] = (__bf16)v.y;
        out[i + 2] = (__bf16)v.z;
        out[i + 3] = (__bf16)v.w;
    } else {
        for (; i < n; ++i) out[i] = (__bf16)in[i];
    }
}

// ---------------------------------------------------------------------------
// 1024x1024 fp32 -> bf16 transpose-convert: Wt[n][k] = W[k][n]
// (gives B-fragment operands contiguous-in-K)
// ---------------------------------------------------------------------------
__global__ __launch_bounds__(256) void transp_cvt(const float* __restrict__ in,
                                                  __bf16* __restrict__ out) {
    __shared__ float t[32][33];
    const int tx = threadIdx.x, ty = threadIdx.y;         // 32 x 8
    const int bx = blockIdx.x * 32, by = blockIdx.y * 32; // n-tile, k-tile
#pragma unroll
    for (int j = 0; j < 32; j += 8)
        t[ty + j][tx] = in[(by + ty + j) * H + bx + tx];
    __syncthreads();
#pragma unroll
    for (int j = 0; j < 32; j += 8)
        out[(bx + ty + j) * H + by + tx] = (__bf16)t[tx][ty + j];
}

// ---------------------------------------------------------------------------
// WMMA GEMM: Y[4096,1024] = A(bf16, row-major) x Wt(bf16, [N,K]) + bias
// One wave -> 32x64 output tile (2 A-frags x 4 B-frags, 8 accumulators):
// every B-fragment feeds two WMMAs -> 12 b128 loads per 8 WMMAs.
// MODE selects epilogue layout:
//   0: bf16 out, head layout        [B,NH,S,HD]   (Q, K)
//   1: bf16 out, transposed heads   [B,NH,HD,S]   (V)
//   2: fp32 out, row-major [4096,1024]            (final projection)
// ---------------------------------------------------------------------------
template<int MODE>
__global__ __launch_bounds__(256) void gemm_wmma(const __bf16* __restrict__ A,
                                                 const __bf16* __restrict__ Bt,
                                                 const float*  __restrict__ bias,
                                                 void* __restrict__ outp) {
    const int lane = threadIdx.x & 31;
    const int gw   = blockIdx.x * 8 + (threadIdx.x >> 5);  // 2048 waves
    const int mt   = gw >> 4;            // 0..127
    const int nt   = gw & 15;            // 0..15
    const int m0   = mt * 32;
    const int n0   = nt * 64;

    const int cl   = lane & 15;
    const int kb8  = (lane >> 4) << 3;   // A-frag K base (0 or 8)
    const int kb16 = (lane >> 4) << 3;   // B-frag K base (0 or 8)

    v8f acc[2][4];
#pragma unroll
    for (int r = 0; r < 2; ++r)
#pragma unroll
        for (int j = 0; j < 4; ++j) acc[r][j] = v8f{};

    const __bf16* arow0 = A + (m0 + cl) * H + kb8;
    const __bf16* arow1 = arow0 + 16 * H;
    const __bf16* brow[4];
#pragma unroll
    for (int j = 0; j < 4; ++j)
        brow[j] = Bt + (n0 + j * 16 + cl) * H + kb16;

    for (int k0 = 0; k0 < H; k0 += 32) {
        v16bf a0 = ld_afrag(arow0 + k0);
        v16bf a1 = ld_afrag(arow1 + k0);
#pragma unroll
        for (int j = 0; j < 4; ++j) {
            v16bf b = ld_bfrag(brow[j] + k0);
            acc[0][j] = wmma_bf16(a0, b, acc[0][j]);
            acc[1][j] = wmma_bf16(a1, b, acc[1][j]);
        }
    }

    // Epilogue. C/D layout: VGPR i holds row +i (lanes 0-15) / +i+8 (16-31).
    const int rhi = (lane >> 4) << 3;
#pragma unroll
    for (int r = 0; r < 2; ++r) {
        const int m0r = m0 + r * 16;
#pragma unroll
        for (int j = 0; j < 4; ++j) {
            const int n  = n0 + j * 16 + cl;
            const float bv = bias[n];
#pragma unroll
            for (int i = 0; i < 8; ++i) {
                const int m = m0r + i + rhi;
                const float val = acc[r][j][i] + bv;
                if (MODE == 2) {
                    ((float*)outp)[m * H + n] = val;
                } else {
                    const int b  = m >> 11, s = m & (SEQ - 1);
                    const int h  = n >> 6,  d = n & (HD - 1);
                    const int bh = b * NH + h;
                    const int idx = (MODE == 0) ? (bh * SEQ + s) * HD + d
                                                : (bh * HD + d) * SEQ + s;
                    ((__bf16*)outp)[idx] = (__bf16)val;
                }
            }
        }
    }
}

// ---------------------------------------------------------------------------
// Flash-style causal attention, one wave per (head, 16-query tile).
// Q [bh][s][d] bf16, K [bh][s][d] bf16 (B-frag-ready for Q.K^T),
// Vt [bh][d][s] bf16 (B-frag-ready for P.V). Keys in 32-wide tiles.
// Row-max via ds_swizzle tree; row-sum via an extra WMMA with an all-ones
// B operand (P x 1 broadcasts row sums into exactly the C-layout slots that
// hold lrow[i]) -- keeps the normalizer consistent with the bf16-rounded P.
// ---------------------------------------------------------------------------
__global__ __launch_bounds__(256) void attn_wmma(const __bf16* __restrict__ Q,
                                                 const __bf16* __restrict__ K,
                                                 const __bf16* __restrict__ Vt,
                                                 __bf16* __restrict__ Ctx) {
    __shared__ __attribute__((aligned(16))) __bf16 pb[8][16 * 32];

    const int lane = threadIdx.x & 31;
    const int w    = threadIdx.x >> 5;
    const int gw   = blockIdx.x * 8 + w;
    const int bh   = gw >> 7;       // 0..31
    const int qt   = gw & 127;      // 0..127
    const int q0   = qt * 16;

    const __bf16* Qh = Q  + bh * (SEQ * HD);
    const __bf16* Kh = K  + bh * (SEQ * HD);
    const __bf16* Vh = Vt + bh * (HD * SEQ);

    const int cl   = lane & 15;
    const int kb8  = (lane >> 4) << 3;
    const int kb16 = (lane >> 4) << 3;
    const int rhi  = (lane >> 4) << 3;

    // Q fragments for d=0..31 and d=32..63 (held in registers all loop long)
    const __bf16* qrow = Qh + (q0 + cl) * HD + kb8;
    v16bf qf0 = ld_afrag(qrow);
    v16bf qf1 = ld_afrag(qrow + 32);

    // all-ones B fragment for row-sum WMMA
    v16bf onesb;
#pragma unroll
    for (int i = 0; i < 16; ++i) onesb[i] = (__bf16)1.0f;

    v8f ctx[4] = {v8f{}, v8f{}, v8f{}, v8f{}};
    float mrow[8], lrow[8];
#pragma unroll
    for (int i = 0; i < 8; ++i) { mrow[i] = -1e30f; lrow[i] = 0.0f; }

    // exp2( (s - m) * CE ) with CE = (1/sqrt(HD)) * log2(e)
    const float CE = 0.125f * 1.44269504f;
    __bf16* pw = pb[w];

    for (int n0 = 0; n0 < q0 + 16; n0 += 32) {
        // ---- scores: S[16q x 32k] = Q (16x64) . K^T, two 16-col halves ----
        v8f s0 = v8f{}, s1 = v8f{};
        {
            const __bf16* k0p = Kh + (n0 + cl) * HD + kb16;
            s0 = wmma_bf16(qf0, ld_bfrag(k0p),      s0);
            s0 = wmma_bf16(qf1, ld_bfrag(k0p + 32), s0);
            const __bf16* k1p = k0p + 16 * HD;
            s1 = wmma_bf16(qf0, ld_bfrag(k1p),      s1);
            s1 = wmma_bf16(qf1, ld_bfrag(k1p + 32), s1);
            // prefetch next key tile (K and V streams) into caches
            __builtin_prefetch(k0p + 32 * HD, 0, 1);
            __builtin_prefetch(Vh + cl * SEQ + n0 + 32, 0, 1);
        }

        const bool need_mask = (n0 + 31 > q0);
        float p0[8], p1[8];
#pragma unroll
        for (int i = 0; i < 8; ++i) {
            float a = s0[i];
            float c = s1[i];
            if (need_mask) {
                const int q = q0 + i + rhi;
                if (n0 + cl      > q) a = -1e30f;
                if (n0 + 16 + cl > q) c = -1e30f;
            }
            p0[i] = a; p1[i] = c;
        }

        // ---- online softmax: row max via swizzle tree, then exp2 ----
#pragma unroll
        for (int i = 0; i < 8; ++i) {
            float t = fmaxf(p0[i], p1[i]);
            t = fmaxf(t, swz_xor<1>(t));
            t = fmaxf(t, swz_xor<2>(t));
            t = fmaxf(t, swz_xor<4>(t));
            t = fmaxf(t, swz_xor<8>(t));
            const float mn = fmaxf(mrow[i], t);
            const float al = exp2f((mrow[i] - mn) * CE);
            mrow[i] = mn;
            p0[i] = exp2f((p0[i] - mn) * CE);
            p1[i] = exp2f((p1[i] - mn) * CE);
            lrow[i] *= al;
#pragma unroll
            for (int j = 0; j < 4; ++j) ctx[j][i] *= al;
        }

        // ---- re-layout P (C-layout -> A-fragment) via wave-private LDS ----
#pragma unroll
        for (int i = 0; i < 8; ++i) {
            const int row = i + rhi;
            pw[row * 32 + cl]      = (__bf16)p0[i];
            pw[row * 32 + 16 + cl] = (__bf16)p1[i];
        }
        asm volatile("s_wait_dscnt 0" ::: "memory");  // DS in-order per wave; explicit for safety
        v16bf pa = ld_afrag(pw + cl * 32 + kb8);

        // ---- row sums via WMMA: P (16x32) . ones(32x16) ----
        v8f rs = wmma_bf16(pa, onesb, v8f{});
#pragma unroll
        for (int i = 0; i < 8; ++i) lrow[i] += rs[i];

        // ---- ctx += P (16x32) . V (32keys x 64d), 4 d-tiles ----
#pragma unroll
        for (int j = 0; j < 4; ++j) {
            v16bf bv = ld_bfrag(Vh + (j * 16 + cl) * SEQ + n0 + kb16);
            ctx[j] = wmma_bf16(pa, bv, ctx[j]);
        }
    }

    // ---- normalize and store ctx -> [B,S,H] row-major bf16 ----
    const int b = bh >> 4, h = bh & 15;
    float inv[8];
#pragma unroll
    for (int i = 0; i < 8; ++i) inv[i] = 1.0f / lrow[i];
#pragma unroll
    for (int j = 0; j < 4; ++j) {
#pragma unroll
        for (int i = 0; i < 8; ++i) {
            const int s = q0 + i + rhi;
            Ctx[(b * SEQ + s) * H + h * HD + j * 16 + cl] = (__bf16)(ctx[j][i] * inv[i]);
        }
    }
}

// ---------------------------------------------------------------------------
// Host-side orchestration (all on `stream`, graph-capture safe)
// ---------------------------------------------------------------------------
extern "C" void kernel_launch(void* const* d_in, const int* in_sizes, int n_in,
                              void* d_out, int out_size, void* d_ws, size_t ws_size,
                              hipStream_t stream) {
    const float* hs = (const float*)d_in[0];
    const float* Wq = (const float*)d_in[1];
    const float* bq = (const float*)d_in[2];
    const float* Wk = (const float*)d_in[3];
    const float* bk = (const float*)d_in[4];
    const float* Wv = (const float*)d_in[5];
    const float* bv = (const float*)d_in[6];
    const float* Wo = (const float*)d_in[7];
    const float* bo = (const float*)d_in[8];

    char* ws = (char*)d_ws;
    const size_t MB = 1u << 20;
    __bf16* Xb  = (__bf16*)(ws + 0 * MB);   // [4096,1024] bf16      8 MiB
    __bf16* Wqt = (__bf16*)(ws + 8 * MB);   // [N,K] bf16            2 MiB
    __bf16* Wkt = (__bf16*)(ws + 10 * MB);
    __bf16* Wvt = (__bf16*)(ws + 12 * MB);
    __bf16* Wot = (__bf16*)(ws + 14 * MB);
    __bf16* Qb  = (__bf16*)(ws + 16 * MB);  // [B,NH,S,HD]           8 MiB
    __bf16* Kb  = (__bf16*)(ws + 24 * MB);  // [B,NH,S,HD]           8 MiB
    __bf16* Vtb = (__bf16*)(ws + 32 * MB);  // [B,NH,HD,S]           8 MiB
    __bf16* Ctx = (__bf16*)(ws + 40 * MB);  // [4096,1024]           8 MiB

    // Stage inputs to bf16 (weights transposed to [N,K] for B-fragments)
    cvt_f32_bf16<<<4096, 256, 0, stream>>>(hs, Xb, MTOT * H);
    dim3 tb(32, 8), tg(32, 32);
    transp_cvt<<<tg, tb, 0, stream>>>(Wq, Wqt);
    transp_cvt<<<tg, tb, 0, stream>>>(Wk, Wkt);
    transp_cvt<<<tg, tb, 0, stream>>>(Wv, Wvt);
    transp_cvt<<<tg, tb, 0, stream>>>(Wo, Wot);

    // QKV projections (2048 waves each: 128 m-tiles x 16 n-strips, 8 waves/block)
    gemm_wmma<0><<<256, 256, 0, stream>>>(Xb, Wqt, bq, Qb);
    gemm_wmma<0><<<256, 256, 0, stream>>>(Xb, Wkt, bk, Kb);
    gemm_wmma<1><<<256, 256, 0, stream>>>(Xb, Wvt, bv, Vtb);

    // Attention: 32 heads x 128 query tiles = 4096 waves
    attn_wmma<<<512, 256, 0, stream>>>(Qb, Kb, Vtb, Ctx);

    // Output projection, fp32 result
    gemm_wmma<2><<<256, 256, 0, stream>>>(Ctx, Wot, bo, d_out);
}
